// GRUModel_20529943675477
// MI455X (gfx1250) — matrix operand order repaired and checked by reading, not compiled, not run on hardware
//
#include <hip/hip_runtime.h>
#include <hip/hip_bf16.h>

// ---------- CDNA5 WMMA types ----------
typedef __attribute__((ext_vector_type(16))) __bf16          v16bf;
typedef __attribute__((ext_vector_type(8)))  float           v8f;
typedef __attribute__((ext_vector_type(8)))  unsigned short  ushort8;
typedef __attribute__((ext_vector_type(4)))  float           f32x4;

static __device__ __forceinline__ unsigned short bf16_rne(float f) {
    unsigned u = __builtin_bit_cast(unsigned, f);
    u += 0x7FFFu + ((u >> 16) & 1u);            // round-to-nearest-even
    return (unsigned short)(u >> 16);
}

// A fragment (16x32 bf16) from a bf16 row: per-lane row m=l%16,
// K-chunks [k0+16*half, +8) and [k0+16*half+8, +8)  -> two 16B loads
static __device__ __forceinline__ v16bf afrag_from_bf16(const unsigned short* row, int k0, int half) {
    union { struct { ushort8 lo, hi; } p; v16bf v; } u;
    u.p.lo = *(const ushort8*)(row + k0 + 16 * half);
    u.p.hi = *(const ushort8*)(row + k0 + 16 * half + 8);
    return u.v;
}

// B fragment (32x16 bf16) from transposed weights W[n][k]: 16 contiguous K at k0+16*half
static __device__ __forceinline__ v16bf bfrag_load(const unsigned short* p) {
    union { struct { ushort8 lo, hi; } p; v16bf v; } u;
    u.p.lo = *(const ushort8*)(p);
    u.p.hi = *(const ushort8*)(p + 8);
    return u.v;
}

static __device__ __forceinline__ v8f wmma_bf16(v16bf a, v16bf b, v8f c) {
    return __builtin_amdgcn_wmma_f32_16x16x32_bf16(false, a, false, b, (short)0, c, false, false);
}

// ---------- Kernel 0: x f32 -> bf16 row-major (bandwidth-bound, 8 elem/thread) ----------
__global__ void __launch_bounds__(256)
x_to_bf16(const float* __restrict__ in, unsigned short* __restrict__ out) {
    const size_t base = ((size_t)blockIdx.x * 256 + threadIdx.x) * 8;   // total 256*128*512
    f32x4 a = *(const f32x4*)(in + base);
    f32x4 b = *(const f32x4*)(in + base + 4);
    ushort8 r;
#pragma unroll
    for (int e = 0; e < 4; ++e) { r[e] = bf16_rne(a[e]); r[4 + e] = bf16_rne(b[e]); }
    *(ushort8*)(out + base) = r;
}

// ---------- Kernel 1: transpose + convert 512x1536 f32 -> [1536][512] bf16 ----------
__global__ void __launch_bounds__(256)
wt_transpose_bf16(const float* __restrict__ in, unsigned short* __restrict__ out) {
    int idx = blockIdx.x * 256 + threadIdx.x;       // over 1536*512
    if (idx >= 512 * 1536) return;
    int n = idx >> 9;          // 0..1535
    int k = idx & 511;         // 0..511
    out[(size_t)n * 512 + k] = bf16_rne(in[(size_t)k * 1536 + n]);
}

// ---------- Kernel 2: xp[t][b][g] = x[b][t][:] @ kernel + bi ----------
// grid: (256 = b, 12 = N-block of 128). block: 256 thr = 8 waves as (wm=4) x (wn=2).
// wave tile: 2 M-subtiles (32 t-rows) x 4 N-subtiles (64 cols) = 8 WMMAs / k-step.
__global__ void __launch_bounds__(256)
xproj_gemm(const unsigned short* __restrict__ xb, const unsigned short* __restrict__ kT,
           const float* __restrict__ bi, float* __restrict__ xp) {
    const int b    = blockIdx.x;
    const int wgn0 = blockIdx.y * 128;
    const int w    = threadIdx.x >> 5;
    const int wm   = w & 3;          // M group: t = wm*32 .. +31
    const int wn   = w >> 2;         // N group: 64 cols
    const int lane = threadIdx.x & 31;
    const int half = lane >> 4;
    const int nl   = lane & 15;

    const int n0 = wgn0 + wn * 64;
    const unsigned short* xrow0 = xb + ((size_t)b * 128 + wm * 32 + nl) * 512;
    const unsigned short* xrow1 = xrow0 + (size_t)16 * 512;

    v8f acc[2][4] = {};
    for (int k0 = 0; k0 < 512; k0 += 32) {
        v16bf a0 = afrag_from_bf16(xrow0, k0, half);
        v16bf a1 = afrag_from_bf16(xrow1, k0, half);
#pragma unroll
        for (int nt = 0; nt < 4; ++nt) {
            v16bf bf = bfrag_load(kT + (size_t)(n0 + nt * 16 + nl) * 512 + k0 + 16 * half);
            acc[0][nt] = wmma_bf16(a0, bf, acc[0][nt]);
            acc[1][nt] = wmma_bf16(a1, bf, acc[1][nt]);
        }
    }
#pragma unroll
    for (int mt = 0; mt < 2; ++mt) {
#pragma unroll
        for (int nt = 0; nt < 4; ++nt) {
            const int n = n0 + nt * 16 + nl;
            const float bias = bi[n];
#pragma unroll
            for (int i = 0; i < 8; ++i) {
                const int t = wm * 32 + mt * 16 + i + 8 * half;
                xp[((size_t)t * 256 + b) * 1536 + n] = acc[mt][nt][i] + bias;
            }
        }
    }
}

// ---------- Kernel 3: fused GRU scan + dense head ----------
// 16 blocks x 256 threads; block owns batch rows [16*blockIdx.x, +16).
// LDS: hF 16x512 f32 (32KB) | hB 16x512 bf16 (16KB) | rec 16x1536 f32 (96KB)
__global__ void __launch_bounds__(256)
gru_scan(const float* __restrict__ xp, const unsigned short* __restrict__ rT,
         const float* __restrict__ br, const float* __restrict__ Wd,
         const float* __restrict__ bd, float* __restrict__ out) {
    extern __shared__ char smem[];
    float*          hF  = (float*)smem;                                  // 16*512 f32
    unsigned short* hB  = (unsigned short*)(smem + 16 * 512 * 4);        // 16*512 bf16
    float*          rec = (float*)(smem + 16 * 512 * 4 + 16 * 512 * 2);  // 16*1536 f32

    const int brow0 = blockIdx.x * 16;
    const int tid   = threadIdx.x;
    const int w     = tid >> 5;
    const int lane  = tid & 31;
    const int half  = lane >> 4;
    const int nl    = lane & 15;

    for (int i = tid; i < 16 * 512; i += 256) { hF[i] = 0.0f; hB[i] = 0; }
    __syncthreads();

    const unsigned short* hrow = hB + (size_t)nl * 512;   // A row m = nl

    for (int t = 0; t < 128; ++t) {
        // Prefetch this step's xp slice (16 contiguous rows = 96KB) so the HBM
        // reads overlap the WMMA phase. 768 lines of 128B over 256 threads.
        const char* xslab = (const char*)(xp + ((size_t)t * 256 + brow0) * 1536);
#pragma unroll
        for (int j = 0; j < 3; ++j)
            __builtin_prefetch(xslab + ((size_t)tid * 3 + j) * 128, 0, 3);

        // ---- rec = h @ Wr : wave w owns n-tiles [w*192, w*192+192) ----
        v8f acc[12] = {};
        for (int k0 = 0; k0 < 512; k0 += 32) {
            v16bf a = afrag_from_bf16(hrow, k0, half);
#pragma unroll
            for (int tl = 0; tl < 12; ++tl) {
                const int n = w * 192 + tl * 16 + nl;
                acc[tl] = wmma_bf16(a, bfrag_load(rT + (size_t)n * 512 + k0 + 16 * half), acc[tl]);
            }
        }
#pragma unroll
        for (int tl = 0; tl < 12; ++tl) {
            const int n = w * 192 + tl * 16 + nl;
#pragma unroll
            for (int i = 0; i < 8; ++i)
                rec[(i + 8 * half) * 1536 + n] = acc[tl][i];
        }
        __syncthreads();

        // ---- gates + state update (f32) ----
        const float* xpt = xp + (size_t)t * 256 * 1536;
        for (int idx = tid; idx < 16 * 512; idx += 256) {
            const int row = idx >> 9;
            const int col = idx & 511;
            const float* xr = xpt + (size_t)(brow0 + row) * 1536;
            const float rz = rec[row * 1536 + col]        + br[col];
            const float rr = rec[row * 1536 + 512 + col]  + br[512 + col];
            const float rh = rec[row * 1536 + 1024 + col] + br[1024 + col];
            const float z  = 1.0f / (1.0f + __expf(-(xr[col] + rz)));
            const float r  = 1.0f / (1.0f + __expf(-(xr[512 + col] + rr)));
            float cand = xr[1024 + col] + r * rh;
            cand = cand > 0.0f ? cand : 0.0f;
            const float hn = z * hF[idx] + (1.0f - z) * cand;
            hF[idx] = hn;
            hB[idx] = bf16_rne(hn);
        }
        __syncthreads();
    }

    // ---- dense head: out[b] = hT[b,:] . Wd + bd ----
    {
        const int row = tid >> 4, seg = tid & 15;
        float p = 0.0f;
        const float* hr = hF + (size_t)row * 512;
#pragma unroll 4
        for (int k = seg * 32; k < seg * 32 + 32; ++k) p += hr[k] * Wd[k];
        rec[row * 16 + seg] = p;            // reuse rec LDS as reduction scratch
        __syncthreads();
        if (tid < 16) {
            float s = 0.0f;
#pragma unroll
            for (int j = 0; j < 16; ++j) s += rec[tid * 16 + j];
            out[brow0 + tid] = s + bd[0];
        }
    }
}

// ---------- host ----------
extern "C" void kernel_launch(void* const* d_in, const int* in_sizes, int n_in,
                              void* d_out, int out_size, void* d_ws, size_t ws_size,
                              hipStream_t stream) {
    const float* x    = (const float*)d_in[0];   // [256][128][512]
    const float* Wk   = (const float*)d_in[1];   // [512][1536]
    const float* Wr   = (const float*)d_in[2];   // [512][1536]
    const float* bias = (const float*)d_in[3];   // [2][1536]
    const float* Wd   = (const float*)d_in[4];   // [512][1]
    const float* bd   = (const float*)d_in[5];   // [1]
    float* out = (float*)d_out;                  // [256]

    char* ws = (char*)d_ws;
    unsigned short* kT = (unsigned short*)ws;                              // 1536*512 bf16
    unsigned short* rT = (unsigned short*)(ws + (size_t)1536 * 512 * 2);   // 1536*512 bf16
    unsigned short* xb = (unsigned short*)(ws + (size_t)2 * 1536 * 512 * 2); // 256*128*512 bf16
    float*          xp = (float*)(ws + (size_t)2 * 1536 * 512 * 2
                                     + (size_t)256 * 128 * 512 * 2);       // [128][256][1536] f32

    const int cvtBlocks = (512 * 1536 + 255) / 256;
    wt_transpose_bf16<<<cvtBlocks, 256, 0, stream>>>(Wk, kT);
    wt_transpose_bf16<<<cvtBlocks, 256, 0, stream>>>(Wr, rT);

    const int xBlocks = (256 * 128 * 512) / (256 * 8);   // 8192
    x_to_bf16<<<xBlocks, 256, 0, stream>>>(x, xb);

    xproj_gemm<<<dim3(256, 12), 256, 0, stream>>>(xb, kT, bias, xp);

    const size_t lds = 16 * 512 * 4 + 16 * 512 * 2 + 16 * 1536 * 4;   // 144 KB < 320 KB/WGP
    gru_scan<<<16, 256, lds, stream>>>(xp, rT, bias + 1536, Wd, bd, out);
}
